// Model_39676907880868
// MI455X (gfx1250) — compile-verified, matrix operand order checked by reading, not run on hardware
//
#include <hip/hip_runtime.h>

typedef __attribute__((ext_vector_type(16))) _Float16 v16h;
typedef __attribute__((ext_vector_type(8)))  float    v8f;

#define S_LEN   2048
#define D_DIM   64
#define NTILE   64
#define WAVES   8
#define BLOCK_M (WAVES * 16)
#define LSTR    68            // padded f16 stride to avoid LDS bank conflicts

// flash attention: D = softmax(Q K^T / s) V, fp32 in/out, f16 WMMA math
__global__ __launch_bounds__(256, 1)
void fattn_wmma_kernel(const float* __restrict__ Q,
                       const float* __restrict__ K,
                       const float* __restrict__ V,
                       const int*   __restrict__ inv_scale,
                       float* __restrict__ O)
{
    __shared__ _Float16 kbuf[NTILE * LSTR];        // [key_row][d]   (B operand of Q*K^T)
    __shared__ _Float16 vbufT[D_DIM * LSTR];       // [d][key_row]   (B operand of P*V)
    __shared__ _Float16 pbuf[WAVES * 16 * LSTR];   // per-wave P tile [m][key_col]

    const int tid  = threadIdx.x;
    const int lane = tid & 31;
    const int wave = tid >> 5;
    const int lm   = lane & 15;     // N index (C layout) / M row (A layout)
    const int half = lane >> 4;     // lane half selects K-pattern offset

    const int nqblk = S_LEN / BLOCK_M;          // 16
    const int bh    = blockIdx.x / nqblk;       // fused (b,h)
    const int qblk  = blockIdx.x % nqblk;

    const float scale = 1.0f / (float)inv_scale[0];
    const size_t base = (size_t)bh * S_LEN * D_DIM;

    // ---- load this wave's Q strip straight into WMMA A fragments (f32 -> f16)
    const int    qrowA = qblk * BLOCK_M + wave * 16 + lm;
    const float* qptr  = Q + base + (size_t)qrowA * D_DIM;
    v16h a0, a1;
#pragma unroll
    for (int j = 0; j < 16; ++j) {
        const int d = ((j < 8) ? j : (j + 8)) + half * 8;   // A-layout K pattern
        a0[j] = (_Float16)qptr[d];
        a1[j] = (_Float16)qptr[d + 32];
    }

    // ---- online-softmax state (per lane: rows M = r + 8*half)
    float m_run[8], l_run[8];
    v8f   oacc[4];
    const v8f vzero = {};
#pragma unroll
    for (int r = 0; r < 8; ++r) { m_run[r] = -1e30f; l_run[r] = 0.0f; }
#pragma unroll
    for (int dt = 0; dt < 4; ++dt) oacc[dt] = vzero;

    for (int nt = 0; nt < S_LEN / NTILE; ++nt) {
        const int k0 = nt * NTILE;

        // ---- cooperative stage: K tile row-major, V tile transposed, fp32 -> f16
#pragma unroll
        for (int it = 0; it < 4; ++it) {
            const int idx = (tid + it * 256) * 4;       // element in 64x64 tile
            const int row = idx >> 6;
            const int d   = idx & 63;
            const size_t goff = base + (size_t)(k0 + row) * D_DIM + d;
            const float4 k4 = *(const float4*)(K + goff);
            const float4 v4 = *(const float4*)(V + goff);
            _Float16* kp = &kbuf[row * LSTR + d];
            kp[0] = (_Float16)k4.x; kp[1] = (_Float16)k4.y;
            kp[2] = (_Float16)k4.z; kp[3] = (_Float16)k4.w;
            vbufT[(d + 0) * LSTR + row] = (_Float16)v4.x;
            vbufT[(d + 1) * LSTR + row] = (_Float16)v4.y;
            vbufT[(d + 2) * LSTR + row] = (_Float16)v4.z;
            vbufT[(d + 3) * LSTR + row] = (_Float16)v4.w;
            if (nt + 1 < S_LEN / NTILE) {               // global_prefetch_b8 next tile
                __builtin_prefetch(K + goff + (size_t)NTILE * D_DIM, 0, 1);
                __builtin_prefetch(V + goff + (size_t)NTILE * D_DIM, 0, 1);
            }
        }
        __syncthreads();

        // ---- scores S = scale * Q K^T  (4 16-wide n-subtiles, K-dim 64 = 2 WMMAs)
        float sc[4][8];
#pragma unroll
        for (int n0 = 0; n0 < 4; ++n0) {
            const _Float16* kr = &kbuf[(n0 * 16 + lm) * LSTR];
            v16h b0, b1;
#pragma unroll
            for (int j = 0; j < 16; ++j) {
                const int d = ((j < 8) ? j : (j + 8)) + half * 8;
                b0[j] = kr[d];
                b1[j] = kr[d + 32];
            }
            v8f c = {};
            c = __builtin_amdgcn_wmma_f32_16x16x32_f16(false, a0, false, b0,
                                                       (short)0, c, false, false);
            c = __builtin_amdgcn_wmma_f32_16x16x32_f16(false, a1, false, b1,
                                                       (short)0, c, false, false);
#pragma unroll
            for (int r = 0; r < 8; ++r) sc[n0][r] = c[r] * scale;
        }

        // ---- online softmax (row = 16 lanes of one half; xor 1/2/4/8 reduces it)
#pragma unroll
        for (int r = 0; r < 8; ++r) {
            float t = fmaxf(fmaxf(sc[0][r], sc[1][r]), fmaxf(sc[2][r], sc[3][r]));
            t = fmaxf(t, __shfl_xor(t, 1, 32));
            t = fmaxf(t, __shfl_xor(t, 2, 32));
            t = fmaxf(t, __shfl_xor(t, 4, 32));
            t = fmaxf(t, __shfl_xor(t, 8, 32));
            const float mnew  = fmaxf(m_run[r], t);
            const float alpha = __expf(m_run[r] - mnew);
            float rs = 0.0f;
#pragma unroll
            for (int n0 = 0; n0 < 4; ++n0) {
                sc[n0][r] = __expf(sc[n0][r] - mnew);
                rs += sc[n0][r];
            }
            rs += __shfl_xor(rs, 1, 32);
            rs += __shfl_xor(rs, 2, 32);
            rs += __shfl_xor(rs, 4, 32);
            rs += __shfl_xor(rs, 8, 32);
            l_run[r] = l_run[r] * alpha + rs;
            m_run[r] = mnew;
#pragma unroll
            for (int dt = 0; dt < 4; ++dt) oacc[dt][r] *= alpha;
        }

        // ---- P: C layout -> per-wave LDS -> A layout (wave-local, dscnt fence)
        _Float16* pw = &pbuf[wave * 16 * LSTR];
#pragma unroll
        for (int n0 = 0; n0 < 4; ++n0)
#pragma unroll
            for (int r = 0; r < 8; ++r)
                pw[(r + 8 * half) * LSTR + n0 * 16 + lm] = (_Float16)sc[n0][r];

        asm volatile("s_wait_dscnt 0" ::: "memory");

        v16h pa0, pa1;
#pragma unroll
        for (int j = 0; j < 16; ++j) {
            const int kk = ((j < 8) ? j : (j + 8)) + half * 8;
            pa0[j] = pw[lm * LSTR + kk];
            pa1[j] = pw[lm * LSTR + 32 + kk];
        }

        // ---- O += P V   (4 d-tiles x 2 k-steps)
#pragma unroll
        for (int dt = 0; dt < 4; ++dt) {
            const _Float16* vr = &vbufT[(dt * 16 + lm) * LSTR];
            v16h vb0, vb1;
#pragma unroll
            for (int j = 0; j < 16; ++j) {
                const int kk = ((j < 8) ? j : (j + 8)) + half * 8;
                vb0[j] = vr[kk];
                vb1[j] = vr[kk + 32];
            }
            oacc[dt] = __builtin_amdgcn_wmma_f32_16x16x32_f16(false, pa0, false, vb0,
                                                              (short)0, oacc[dt], false, false);
            oacc[dt] = __builtin_amdgcn_wmma_f32_16x16x32_f16(false, pa1, false, vb1,
                                                              (short)0, oacc[dt], false, false);
        }
        __syncthreads();   // protect kbuf/vbuf before next stage
    }

    // ---- epilogue: normalize and store fp32 output
#pragma unroll
    for (int dt = 0; dt < 4; ++dt)
#pragma unroll
        for (int r = 0; r < 8; ++r) {
            const int M = r + 8 * half;
            const size_t orow = base + (size_t)(qblk * BLOCK_M + wave * 16 + M) * D_DIM;
            O[orow + dt * 16 + lm] = oacc[dt][r] / l_run[r];
        }
}

extern "C" void kernel_launch(void* const* d_in, const int* in_sizes, int n_in,
                              void* d_out, int out_size, void* d_ws, size_t ws_size,
                              hipStream_t stream) {
    (void)in_sizes; (void)n_in; (void)out_size; (void)d_ws; (void)ws_size;
    const float* Q  = (const float*)d_in[0];
    const float* K  = (const float*)d_in[1];
    const float* V  = (const float*)d_in[2];
    const int*   is = (const int*)  d_in[3];
    float*       O  = (float*)d_out;

    const int B = 4, H = 16;
    const int blocks = B * H * (S_LEN / BLOCK_M);   // 1024
    fattn_wmma_kernel<<<blocks, 256, 0, stream>>>(Q, K, V, is, O);
}
